// MetaRDU_13245679141240
// MI455X (gfx1250) — matrix operand order repaired and checked by reading, not compiled, not run on hardware
//
#include <hip/hip_runtime.h>
#include <hip/hip_bf16.h>

// ---------------------------------------------------------------------------
// MetaRDU on gfx1250. All GEMM-shaped work (3x3/9x9/2x2s2 convs, deconv
// sub-GEMMs, Pos2Weight GEMM) -> one implicit-GEMM kernel on
// v_wmma_f32_16x16x32_f16 (wave32). ~113 G MACs => compute bound on MI455X
// (~0.5 GB traffic ~= 22us at 23.3 TB/s). Weight tiles are moved by the
// Tensor Data Mover (tensor_load_to_lds + s_wait_tensorcnt), overlapping the
// VALU im2col gather; TDM's tensor_dim OOB-zeroing and LDS row padding
// implement the tile guards and the bank-conflict pitch in hardware.
// ---------------------------------------------------------------------------

typedef __attribute__((ext_vector_type(16))) _Float16 v16h;
typedef __attribute__((ext_vector_type(8)))  float    v8f;
typedef __attribute__((ext_vector_type(4)))  unsigned uint4v;
typedef __attribute__((ext_vector_type(8)))  int      int8v;
typedef __attribute__((ext_vector_type(4)))  int      int4v;

#define MT 64    // Cout tile per workgroup (4 wmma m-tiles)
#define NT 128   // pixel tile per workgroup (8 wmma n-tiles)
#define KT 32    // K step == wmma K

#if __has_include(<hip/amd_detail/amd_gfx1250_TDM.h>)
#define TDM_SIX_ARGS 1
#else
#define TDM_SIX_ARGS 0
#endif

// ---------------------------------------------------------------------------
// TDM 2D tile load: KT x MT tile of 2-byte elements into LDS, row pitch
// padded by 1 dword every 16 dwords (64B row -> 68B pitch == KT+2 halves).
// tensor_dim0/1 = remaining extents so out-of-range rows/cols read as zero.
// ---------------------------------------------------------------------------
__device__ inline void tdm_load_tileA(const _Float16* gsrc, unsigned ldsAddr,
                                      int rowsRemain, int colsRemain, int rowStride)
{
  unsigned long long ga = (unsigned long long)gsrc;
  unsigned td0 = (unsigned)(colsRemain < 0 ? 0 : colsRemain);   // K extent
  unsigned td1 = (unsigned)(rowsRemain < 0 ? 0 : rowsRemain);   // Cout extent

  uint4v g0;
  g0[0] = 1u;                                            // count=1 (valid user D#)
  g0[1] = ldsAddr;                                       // lds_addr
  g0[2] = (unsigned)(ga & 0xFFFFFFFFu);                  // global_addr[31:0]
  g0[3] = (unsigned)((ga >> 32) & 0x01FFFFFFu)           // global_addr[56:32]
        | (2u << 30);                                    // type=2 ("image")

  int8v g1;
  g1[0] = (int)((1u << 16) |                             // data_size = 2 bytes
                (1u << 20) |                             // pad_enable
                (3u << 22));                             // pad_interval = 16 dwords
                                                         // pad_amount = 0 -> 1 dword
  g1[1] = (int)((td0 & 0xFFFFu) << 16);                  // tensor_dim0[15:0]
  g1[2] = (int)(((td0 >> 16) & 0xFFFFu) |                // tensor_dim0[31:16]
                ((td1 & 0xFFFFu) << 16));                // tensor_dim1[15:0]
  g1[3] = (int)(((td1 >> 16) & 0xFFFFu) |                // tensor_dim1[31:16]
                ((unsigned)KT << 16));                   // tile_dim0 = KT
  g1[4] = (int)MT;                                       // tile_dim1 = MT, tile_dim2=0
  g1[5] = rowStride;                                     // tensor_dim0_stride[31:0]
  g1[6] = 0;                                             // stride hi / dim1 stride lo
  g1[7] = 0;

  int4v gz = {0, 0, 0, 0};
#if TDM_SIX_ARGS
  int8v gz8 = {0, 0, 0, 0, 0, 0, 0, 0};
  __builtin_amdgcn_tensor_load_to_lds(g0, g1, gz, gz, gz8, 0);
#else
  __builtin_amdgcn_tensor_load_to_lds(g0, g1, gz, gz, 0);
#endif
}

// ---------------------------------------------------------------------------
// Generic implicit-GEMM conv:  out[m, p] = act( sum_k W[m,k] * im2col[k,p] )
//   k = ci*KH*KW + kh*KW + kw ; p = oy*Wout + ox
// Workgroup: 256 threads = 8 waves; tile 64(M) x 128(N); each wave holds a
// 16x64 strip = 4 accumulators, reusing one A fragment across 4 WMMAs/K-step.
// Epilogue: +bias, PReLU(alpha), +residual (residPlane==0 -> broadcast),
// scatter-capable output addressing (ConvTranspose2d sub-pixel writes).
// ---------------------------------------------------------------------------
__global__ __launch_bounds__(256) void conv_wmma(
    const _Float16* __restrict__ in, const _Float16* __restrict__ wA,
    const float* __restrict__ bias, const float* __restrict__ alpha,
    const _Float16* __restrict__ resid, int residPlane,
    _Float16* __restrict__ out, int outPlane, int outRow, int outCol, int outOff,
    int Cin, int Cout, int Hin, int Win, int Hout, int Wout,
    int KH, int KW, int stride, int pad)
{
  const int K    = Cin * KH * KW;
  const int Npix = Hout * Wout;
  const int mBase = blockIdx.y * MT;
  const int nBase = blockIdx.x * NT;
  const int tid  = threadIdx.x;
  const int lane = tid & 31;
  const int wave = tid >> 5;
  const int mt = (wave >> 1) << 4;   // 0,16,32,48
  const int nh = (wave & 1) << 6;    // 0 or 64

  __shared__ _Float16 sA[MT][KT + 2];   // [m][k], pitch matches TDM padding
  __shared__ _Float16 sB[NT][KT + 2];   // [pixel][k] (transposed for b32 reads)

  const unsigned ldsA = (unsigned)(unsigned long long)&sA[0][0];

  v8f acc0 = {}, acc1 = {}, acc2 = {}, acc3 = {};
  union U { v16h h; unsigned u[8]; };

  for (int kt = 0; kt < K; kt += KT) {
    // Weight tile via Tensor Data Mover (wave-level op; EXEC ignored).
    if (wave == 0)
      tdm_load_tileA(wA + (long)mBase * K + kt, ldsA, Cout - mBase, K - kt, K);

    // im2col gather of the B tile by all 256 threads, overlapping the TDM.
    for (int i = tid; i < NT * KT; i += 256) {
      int p = i >> 5, kk = i & 31;
      int kg = kt + kk, pg = nBase + p;
      _Float16 v = (_Float16)0.f;
      if (kg < K && pg < Npix) {
        int ci  = kg / (KH * KW);
        int rem = kg - ci * KH * KW;
        int kh  = rem / KW, kw = rem - kh * KW;
        int oy  = pg / Wout, ox = pg - oy * Wout;
        int iy  = oy * stride - pad + kh;
        int ix  = ox * stride - pad + kw;
        if (iy >= 0 && iy < Hin && ix >= 0 && ix < Win)
          v = in[(long)ci * Hin * Win + (long)iy * Win + ix];
      }
      sB[p][kk] = v;
    }
    if (wave == 0) __builtin_amdgcn_s_wait_tensorcnt(0);
    __syncthreads();

    // A fragment (ISA 7.12.2 16-bit A layout), packed 32-bit LDS reads.
    U a;
    const unsigned* rowA = (const unsigned*)sA[mt + (lane & 15)];
    const int kA = (lane >> 4) << 3;     // 0 or 8
#pragma unroll
    for (int e = 0; e < 8; ++e) {
      int k = ((e >> 2) << 4) + kA + ((e & 3) << 1);
      a.u[e] = rowA[k >> 1];
    }

    const int kB2 = (lane >> 4) << 2;    // b32 index of K base (0 or 8)/2
#pragma unroll
    for (int t = 0; t < 4; ++t) {
      U b;
      const unsigned* rowB = (const unsigned*)sB[nh + (t << 4) + (lane & 15)];
#pragma unroll
      for (int e = 0; e < 8; ++e) b.u[e] = rowB[((e >> 2) << 3) + kB2 + (e & 3)];
      v8f& acc = (t == 0 ? acc0 : t == 1 ? acc1 : t == 2 ? acc2 : acc3);
      acc = __builtin_amdgcn_wmma_f32_16x16x32_f16(
          false, a.h, false, b.h, (short)0, acc, false, false);
    }
    __syncthreads();
  }

  // Epilogue: C/D layout VGPR r -> M=r (lanes 0-15) / M=r+8 (lanes 16-31).
  const int n0 = nBase + nh + (lane & 15);
  const int m0 = mBase + mt + ((lane >> 4) << 3);
#pragma unroll
  for (int t = 0; t < 4; ++t) {
    v8f acc = (t == 0 ? acc0 : t == 1 ? acc1 : t == 2 ? acc2 : acc3);
    int pg = n0 + (t << 4);
    if (pg < Npix) {
      int oy = pg / Wout, ox = pg - oy * Wout;
#pragma unroll
      for (int r = 0; r < 8; ++r) {
        int m = m0 + r;
        if (m < Cout) {
          float v = acc[r];
          if (bias)  v += bias[m];
          if (alpha) v = (v >= 0.f) ? v : alpha[m] * v;
          if (resid) v += (float)resid[(long)m * residPlane + pg];
          out[(long)m * outPlane + (long)oy * outRow + (long)ox * outCol + outOff] =
              (_Float16)v;
        }
      }
    }
  }
}

// ---------------------------------------------------------------------------
// Helper kernels (memory-bound / tiny ops)
// ---------------------------------------------------------------------------
__global__ void pack_w_f16(_Float16* __restrict__ dst, const float* __restrict__ src,
                           int M, int Kd, long sM, long sK, long off)
{
  long i = (long)blockIdx.x * 256 + threadIdx.x;
  if (i >= (long)M * Kd) return;
  int m = (int)(i / Kd), k = (int)(i - (long)m * Kd);
  dst[i] = (_Float16)src[off + (long)m * sM + (long)k * sK];
}

__global__ void cvt_f32_f16(_Float16* __restrict__ dst, const float* __restrict__ src, long n)
{
  long i = (long)blockIdx.x * 256 + threadIdx.x;
  if (i < n) dst[i] = (_Float16)src[i];
}

__global__ void copy_f16(_Float16* __restrict__ dst, const _Float16* __restrict__ src, long n)
{
  long i = (long)blockIdx.x * 256 + threadIdx.x;
  if (i < n) dst[i] = src[i];
}

// h = relu(a @ mrm1.w^T + mrm1.b) : 2 -> 128
__global__ void mrm_h(float* __restrict__ h, const float* __restrict__ a,
                      const float* __restrict__ w, const float* __restrict__ b)
{
  int j = threadIdx.x;
  float v = a[0] * w[j * 2 + 0] + a[1] * w[j * 2 + 1] + b[j];
  h[j] = v > 0.f ? v : 0.f;
}

// wk[j] = h . mrm2.w[j,:] + mrm2.b[j]  (147456 outputs, 75MB read: HBM bound)
__global__ void mrm_wk(_Float16* __restrict__ wk, const float* __restrict__ h,
                       const float* __restrict__ w2, const float* __restrict__ b2, int n)
{
  int j = blockIdx.x * 256 + threadIdx.x;
  if (j >= n) return;
  const float* row = w2 + (long)j * 128;
  __builtin_prefetch(row + 32, 0, 0);       // global_prefetch_b8
  __builtin_prefetch(row + 96, 0, 0);
  float acc = b2[j];
#pragma unroll 8
  for (int k = 0; k < 128; ++k) acc += h[k] * row[k];
  wk[j] = (_Float16)acc;
}

// hlT[k, p] = relu(pm[p,:4] . w1[k,:4] + b1[k]) stored channel-major for GEMM
__global__ void p2w_hl(_Float16* __restrict__ hlT, const float* __restrict__ pm,
                       const float* __restrict__ w, const float* __restrict__ b, int HW)
{
  long i = (long)blockIdx.x * 256 + threadIdx.x;
  if (i >= (long)256 * HW) return;
  int k = (int)(i / HW), p = (int)(i - (long)k * HW);
  const float* pr = pm + (long)p * 4;
  float acc = b[k] + pr[0] * w[k * 4 + 0] + pr[1] * w[k * 4 + 1]
                   + pr[2] * w[k * 4 + 2] + pr[3] * w[k * 4 + 3];
  hlT[i] = (_Float16)(acc > 0.f ? acc : 0.f);
}

// out[p] = sum_{c,j} unfold(xs)[c*9+j, p] * lwT[c*9+j, p]
__global__ void dynfilt(float* __restrict__ out, const _Float16* __restrict__ xs,
                        const _Float16* __restrict__ lwT, int H, int W)
{
  int p = blockIdx.x * 256 + threadIdx.x;
  int HW = H * W;
  if (p >= HW) return;
  int py = p / W, px = p - py * W;
  float acc = 0.f;
  for (int c = 0; c < 128; ++c) {
    const _Float16* xc = xs + (long)c * HW;
    int j = 0;
#pragma unroll
    for (int dy = -1; dy <= 1; ++dy)
#pragma unroll
      for (int dx = -1; dx <= 1; ++dx, ++j) {
        int y = py + dy, x = px + dx;
        float xv = (y >= 0 && y < H && x >= 0 && x < W) ? (float)xc[y * W + x] : 0.f;
        acc += xv * (float)lwT[(long)(c * 9 + j) * HW + p];
      }
  }
  out[p] = acc;
}

// ---------------------------------------------------------------------------
// Host-side orchestration
// ---------------------------------------------------------------------------
namespace {

struct Ws { char* p; size_t off; };
static void* wsalloc(Ws& w, size_t bytes) {
  size_t o = (w.off + 255) & ~(size_t)255;
  w.off = o + bytes;
  return w.p + o;
}
static _Float16* wsH(Ws& w, size_t n) { return (_Float16*)wsalloc(w, n * 2); }
static float*    wsF(Ws& w, size_t n) { return (float*)wsalloc(w, n * 4); }

#define F32(q) ((const float*)(q))

static void conv(hipStream_t s, const _Float16* in, const _Float16* wA,
                 const float* bias, const float* alpha,
                 const _Float16* resid, int residPlane,
                 _Float16* out, int outPlane, int outRow, int outCol, int outOff,
                 int Cin, int Cout, int Hin, int Win, int Hout, int Wout,
                 int KH, int KW, int stride, int pad)
{
  dim3 g((Hout * Wout + NT - 1) / NT, (Cout + MT - 1) / MT);
  conv_wmma<<<g, 256, 0, s>>>(in, wA, bias, alpha, resid, residPlane,
                              out, outPlane, outRow, outCol, outOff,
                              Cin, Cout, Hin, Win, Hout, Wout, KH, KW, stride, pad);
}

static _Float16* packW(Ws& ws, hipStream_t s, const float* src,
                       int M, int Kd, long sM, long sK, long off = 0)
{
  _Float16* d = wsH(ws, (size_t)M * Kd);
  long n = (long)M * Kd;
  pack_w_f16<<<(unsigned)((n + 255) / 256), 256, 0, s>>>(d, src, M, Kd, sM, sK, off);
  return d;
}

// Dense block leaves (sorted): a0 a1 a2 a3 c0.b c0.w c1.b c1.w c2.b c2.w c3.b c3.w
static void denoise(hipStream_t s, Ws& ws, void* const* din, int bi,
                    _Float16* cat, _Float16* outb, int Cin, int ci, int co, int H, int W)
{
  const int HW = H * W;
  const float *a0 = F32(din[bi+0]), *a1 = F32(din[bi+1]),
              *a2 = F32(din[bi+2]), *a3 = F32(din[bi+3]);
  const float *c0b = F32(din[bi+4]),  *c0w = F32(din[bi+5]);
  const float *c1b = F32(din[bi+6]),  *c1w = F32(din[bi+7]);
  const float *c2b = F32(din[bi+8]),  *c2w = F32(din[bi+9]);
  const float *c3b = F32(din[bi+10]), *c3w = F32(din[bi+11]);

  int k0 = Cin * 9, k1 = (Cin + ci) * 9, k2 = (Cin + 2 * ci) * 9, k3 = (Cin + 3 * ci) * 9;
  _Float16* w0 = packW(ws, s, c0w, ci, k0, k0, 1);
  conv(s, cat, w0, c0b, a0, nullptr, 0, cat + (size_t)Cin * HW, HW, W, 1, 0,
       Cin, ci, H, W, H, W, 3, 3, 1, 1);
  _Float16* w1 = packW(ws, s, c1w, ci, k1, k1, 1);
  conv(s, cat, w1, c1b, a1, nullptr, 0, cat + (size_t)(Cin + ci) * HW, HW, W, 1, 0,
       Cin + ci, ci, H, W, H, W, 3, 3, 1, 1);
  _Float16* w2 = packW(ws, s, c2w, ci, k2, k2, 1);
  conv(s, cat, w2, c2b, a2, nullptr, 0, cat + (size_t)(Cin + 2 * ci) * HW, HW, W, 1, 0,
       Cin + 2 * ci, ci, H, W, H, W, 3, 3, 1, 1);
  _Float16* w3 = packW(ws, s, c3w, co, k3, k3, 1);
  conv(s, cat, w3, c3b, a3, /*resid=*/cat, HW, outb, HW, W, 1, 0,
       Cin + 3 * ci, co, H, W, H, W, 3, 3, 1, 1);
}

// Down leaves (sorted): a c.b c.w   -- 2x2 stride-2 conv + PReLU
static void down(hipStream_t s, Ws& ws, void* const* din, int di,
                 const _Float16* in, int Cin, int Cout, int h, int w, _Float16* outCat)
{
  const float *a = F32(din[di+0]), *cb = F32(din[di+1]), *cw = F32(din[di+2]);
  _Float16* wD = packW(ws, s, cw, Cout, Cin * 4, (long)Cin * 4, 1);
  conv(s, in, wD, cb, a, nullptr, 0, outCat, (h / 2) * (w / 2), w / 2, 1, 0,
       Cin, Cout, h, w, h / 2, w / 2, 2, 2, 2, 0);
}

// Up leaves (sorted): a at bt c.b c.w wt
// ConvTranspose2d(k=2,s=2) == 4 sub-pixel GEMMs scattered into the upsampled
// plane; then concat [cat, u] and 3x3 conv.
static void up(hipStream_t s, Ws& ws, void* const* din, int ui,
               const _Float16* upin, int Cup, int h, int w,
               const _Float16* cat_src, int Ccat,
               _Float16* ucat, _Float16* outb, int Cout)
{
  const float *a  = F32(din[ui+0]), *at = F32(din[ui+1]), *bt = F32(din[ui+2]);
  const float *cb = F32(din[ui+3]), *cw = F32(din[ui+4]), *wt = F32(din[ui+5]);
  const int H = 2 * h, Wd = 2 * w;
  const long ncat = (long)Ccat * H * Wd;

  copy_f16<<<(unsigned)((ncat + 255) / 256), 256, 0, s>>>(ucat, cat_src, ncat);

  for (int d = 0; d < 2; ++d)
    for (int k = 0; k < 2; ++k) {
      // A[m=o, kk=c] = wt[c, o, d, k]  (codk layout)
      _Float16* wA = packW(ws, s, wt, Cup, Cup, /*sM=*/4, /*sK=*/(long)Cup * 4,
                           /*off=*/d * 2 + k);
      conv(s, upin, wA, bt, at, nullptr, 0,
           ucat + ncat, /*outPlane=*/H * Wd, /*outRow=*/2 * Wd, /*outCol=*/2,
           /*outOff=*/d * Wd + k,
           Cup, Cup, h, w, h, w, 1, 1, 1, 0);
    }

  int kc = (Ccat + Cup) * 9;
  _Float16* wC = packW(ws, s, cw, Cout, kc, kc, 1);
  conv(s, ucat, wC, cb, a, nullptr, 0, outb, H * Wd, Wd, 1, 0,
       Ccat + Cup, Cout, H, Wd, H, Wd, 3, 3, 1, 1);
}

} // namespace

extern "C" void kernel_launch(void* const* d_in, const int* in_sizes, int n_in,
                              void* d_out, int out_size, void* d_ws, size_t ws_size,
                              hipStream_t stream)
{
  (void)in_sizes; (void)n_in; (void)out_size; (void)ws_size;
  // Input order: x, pos_mat, a, then params pytree leaves (dict keys sorted).
  const float* x   = F32(d_in[0]);
  const float* pos = F32(d_in[1]);
  const float* av  = F32(d_in[2]);
  const int PB = 3;
  const int b00 = PB + 0,  b02 = PB + 12, b10 = PB + 24, b12 = PB + 36;
  const int b20 = PB + 48, b22 = PB + 60, b30 = PB + 72;
  const int d0 = PB + 84, d1 = PB + 87, d2 = PB + 90;
  const int in_a1 = PB + 93, in_c1b = PB + 94, in_c1w = PB + 95;
  const int mrm1b = PB + 96,  mrm1w = PB + 97,  mrm2b = PB + 98,  mrm2w = PB + 99;
  const int p2w1b = PB + 100, p2w1w = PB + 101, p2w2b = PB + 102, p2w2w = PB + 103;
  const int u0 = PB + 104, u1 = PB + 110, u2 = PB + 116;

  Ws ws{(char*)d_ws, 0};
  const int H = 128, W = 128, HW = H * W;
  hipStream_t s = stream;

  // x -> f16
  _Float16* xb = wsH(ws, HW);
  cvt_f32_f16<<<(HW + 255) / 256, 256, 0, s>>>(xb, x, HW);

  // InputBlock: 9x9 conv 1->128 (pad 4) + PReLU, written into b00's concat buf
  _Float16* cat0a = wsH(ws, (size_t)320 * HW);
  _Float16* win = packW(ws, s, F32(d_in[in_c1w]), 128, 81, 81, 1);
  conv(s, xb, win, F32(d_in[in_c1b]), F32(d_in[in_a1]), nullptr, 0,
       cat0a, HW, W, 1, 0, 1, 128, H, W, H, W, 9, 9, 1, 4);

  _Float16* o0 = wsH(ws, (size_t)128 * HW);
  denoise(s, ws, d_in, b00, cat0a, o0, 128, 64, 128, H, W);

  _Float16* cat1a = wsH(ws, (size_t)640 * 4096);
  down(s, ws, d_in, d0, o0, 128, 256, 128, 128, cat1a);
  _Float16* o1 = wsH(ws, (size_t)256 * 4096);
  denoise(s, ws, d_in, b10, cat1a, o1, 256, 128, 256, 64, 64);

  _Float16* cat2a = wsH(ws, (size_t)1280 * 1024);
  down(s, ws, d_in, d1, o1, 256, 512, 64, 64, cat2a);
  _Float16* o2 = wsH(ws, (size_t)512 * 1024);
  denoise(s, ws, d_in, b20, cat2a, o2, 512, 256, 512, 32, 32);

  _Float16* cat3 = wsH(ws, (size_t)2560 * 256);
  down(s, ws, d_in, d2, o2, 512, 1024, 32, 32, cat3);
  _Float16* o3 = wsH(ws, (size_t)1024 * 256);
  denoise(s, ws, d_in, b30, cat3, o3, 1024, 512, 1024, 16, 16);

  // Decoder
  _Float16* ucat2 = wsH(ws, (size_t)1536 * 1024);
  _Float16* cat2b = wsH(ws, (size_t)1280 * 1024);
  up(s, ws, d_in, u2, o3, 1024, 16, 16, o2, 512, ucat2, cat2b, 512);
  _Float16* o4 = wsH(ws, (size_t)512 * 1024);
  denoise(s, ws, d_in, b22, cat2b, o4, 512, 256, 512, 32, 32);

  _Float16* ucat1 = wsH(ws, (size_t)768 * 4096);
  _Float16* cat1b = wsH(ws, (size_t)640 * 4096);
  up(s, ws, d_in, u1, o4, 512, 32, 32, o1, 256, ucat1, cat1b, 256);
  _Float16* o5 = wsH(ws, (size_t)256 * 4096);
  denoise(s, ws, d_in, b12, cat1b, o5, 256, 128, 256, 64, 64);

  _Float16* ucat0 = wsH(ws, (size_t)384 * HW);
  _Float16* cat0b = wsH(ws, (size_t)320 * HW);
  up(s, ws, d_in, u0, o5, 256, 64, 64, o0, 128, ucat0, cat0b, 128);
  _Float16* o6 = wsH(ws, (size_t)128 * HW);
  denoise(s, ws, d_in, b02, cat0b, o6, 128, 64, 128, H, W);

  // MRM meta conv: wk = mrm2(relu(mrm1(a))) -> [128, 128*9] f16, then
  // o7 = conv3x3(o6, wk); xs = o7 + x (1-channel broadcast residual)
  float* hbuf = wsF(ws, 128);
  mrm_h<<<1, 128, 0, s>>>(hbuf, av, F32(d_in[mrm1w]), F32(d_in[mrm1b]));
  _Float16* wk = wsH(ws, 147456);
  mrm_wk<<<(147456 + 255) / 256, 256, 0, s>>>(wk, hbuf, F32(d_in[mrm2w]),
                                              F32(d_in[mrm2b]), 147456);
  _Float16* xs = wsH(ws, (size_t)128 * HW);
  conv(s, o6, wk, nullptr, nullptr, /*resid=*/xb, /*residPlane=*/0,
       xs, HW, W, 1, 0, 128, 128, H, W, H, W, 3, 3, 1, 1);

  // Pos2Weight: hlT [256, HW] then lwT = GEMM(p2w2.w [1152x256], hlT) as 1x1 conv
  _Float16* hlT = wsH(ws, (size_t)256 * HW);
  p2w_hl<<<(unsigned)(((size_t)256 * HW + 255) / 256), 256, 0, s>>>(
      hlT, pos, F32(d_in[p2w1w]), F32(d_in[p2w1b]), HW);
  _Float16* wp = packW(ws, s, F32(d_in[p2w2w]), 1152, 256, 256, 1);
  _Float16* lwT = wsH(ws, (size_t)1152 * HW);
  conv(s, hlT, wp, F32(d_in[p2w2b]), nullptr, nullptr, 0,
       lwT, HW, W, 1, 0, 256, 1152, H, W, H, W, 1, 1, 1, 0);

  // Final per-pixel dynamic 3x3 filter
  dynfilt<<<(HW + 255) / 256, 256, 0, s>>>((float*)d_out, xs, lwT, H, W);
}
